// NeuralstackOnly_37512244363492
// MI455X (gfx1250) — matrix operand order repaired and checked by reading, not compiled, not run on hardware
//
#include <hip/hip_runtime.h>
#include <math.h>

#define B_DIM 64
#define S_DIM 256
#define V_DIM 2048
#define N_STACK 12
#define KPAD 288           // S + 1 (latch_init col) padded to multiple of 32
#define ZERO_OFF 0.001f
#define EPS_CS 1e-8f
#define BS (B_DIM * S_DIM)
#define BSV ((size_t)B_DIM * S_DIM * V_DIM)

typedef __attribute__((ext_vector_type(16))) __bf16 v16bf;
typedef __attribute__((ext_vector_type(8)))  float  v8f;

// ---------------- K0: norms of latch_enable / should_pop -------------------
__global__ __launch_bounds__(256) void k_norms(const float* __restrict__ le,
                                               const float* __restrict__ sp,
                                               float* __restrict__ scal) {
  int tid = threadIdx.x;
  float a = 0.f, b = 0.f;
  for (int v = tid; v < V_DIM; v += 256) { float l = le[v]; a += l * l; float s = sp[v]; b += s * s; }
  __shared__ float r1[256], r2[256];
  r1[tid] = a; r2[tid] = b; __syncthreads();
  for (int off = 128; off; off >>= 1) {
    if (tid < off) { r1[tid] += r1[tid + off]; r2[tid] += r2[tid + off]; }
    __syncthreads();
  }
  if (tid == 0) { scal[0] = fmaxf(sqrtf(r1[0]), EPS_CS); scal[1] = fmaxf(sqrtf(r2[0]), EPS_CS); }
}

// ---------------- K1: per (b,s): p = <le,x>, nrm = ||x||^2 -----------------
__global__ __launch_bounds__(256) void k_row_reduce(const float* __restrict__ x,
                                                    const float* __restrict__ le,
                                                    float* __restrict__ p,
                                                    float* __restrict__ nrm) {
  int bs = blockIdx.x, tid = threadIdx.x;
  const float* row = x + (size_t)bs * V_DIM;
  float pp = 0.f, nn = 0.f;
  for (int v = tid; v < V_DIM; v += 256) { float xv = row[v]; pp += le[v] * xv; nn += xv * xv; }
  __shared__ float r1[256], r2[256];
  r1[tid] = pp; r2[tid] = nn; __syncthreads();
  for (int off = 128; off; off >>= 1) {
    if (tid < off) { r1[tid] += r1[tid + off]; r2[tid] += r2[tid + off]; }
    __syncthreads();
  }
  if (tid == 0) { p[bs] = r1[0]; nrm[bs] = r2[0]; }
}

// ---------------- K2a: e, cumprod(1-e), g = e/cp ---------------------------
__global__ __launch_bounds__(256) void k_prefix(const float* __restrict__ p,
                                                const float* __restrict__ nrm,
                                                const float* __restrict__ scal,
                                                float* __restrict__ cp,
                                                float* __restrict__ g) {
  int b = blockIdx.x, s = threadIdx.x;
  __shared__ float se[256], sa[256];
  float nle = scal[0];
  int bs = b * S_DIM + s;
  float e = p[bs] / (nle * fmaxf(sqrtf(nrm[bs]), EPS_CS));
  se[s] = e; sa[s] = 1.f - e;
  __syncthreads();
  float val = sa[s];
  for (int off = 1; off < 256; off <<= 1) {       // Hillis-Steele prefix product
    float prev = (s >= off) ? sa[s - off] : 1.f;
    __syncthreads();
    val *= prev; sa[s] = val;
    __syncthreads();
  }
  float cps = sa[s];
  cp[bs] = cps;
  g[b * KPAD + s] = se[s] / cps;
  if (s == 0) g[b * KPAD + S_DIM] = 1.f;          // latch_init column
  if (s >= 1 && s < 32) g[b * KPAD + S_DIM + s] = 0.f;  // zero pad cols
}

// ---------------- K2b: latch_states = T @ [x; latch_init] via WMMA ---------
// A[s][t] = (t<=s || t>=S) ? cp_s * g_t : 0 ; bf16 hi/lo split, 3 WMMAs/tile.
__global__ __launch_bounds__(256) void latch_gemm_wmma(const float* __restrict__ x,
                                                       const float* __restrict__ latch_init,
                                                       const float* __restrict__ cp,
                                                       const float* __restrict__ g,
                                                       float* __restrict__ latch_out) {
  const int b = blockIdx.z;
  const int sBase = blockIdx.y * 64;
  const int vBase = blockIdx.x * 128;
  const int tid = threadIdx.x;
  const int lane = tid & 31;
  const int wave = tid >> 5;
  const int wS = wave >> 2;        // 0..1 : 32-row slab
  const int wV = wave & 3;         // 0..3 : 32-col slab
  const int h = lane >> 4;         // K-half select per ISA layout
  const int l15 = lane & 15;

  __shared__ __bf16 Ahi[64][32], Alo[64][32];
  __shared__ __bf16 Bhi[128][32], Blo[128][32];   // stored [n][k] for contiguous frag loads

  v8f acc[2][2] = {};
  const float* cpb = cp + b * S_DIM;
  const float* gb  = g  + b * KPAD;

  for (int kk = 0; kk < KPAD; kk += 32) {
    // Stage A tile (64 s-rows x 32 k) computed on the fly from cp/g
    for (int i = tid; i < 64 * 32; i += 256) {
      int ls = i >> 5, lk = i & 31;
      int s = sBase + ls, t = kk + lk;
      float f = 0.f;
      if (t <= s || t >= S_DIM) f = cpb[s] * gb[t];
      __bf16 fh = (__bf16)f;
      Ahi[ls][lk] = fh;
      Alo[ls][lk] = (__bf16)(f - (float)fh);
    }
    // Stage B tile (32 k x 128 v) from x / latch_init, transposed into LDS
    for (int i = tid; i < 32 * 128; i += 256) {
      int lk = i >> 7, lv = i & 127;
      int t = kk + lk;
      float f = 0.f;
      if (t < S_DIM)       f = x[((size_t)b * S_DIM + t) * V_DIM + vBase + lv];
      else if (t == S_DIM) f = latch_init[(size_t)b * V_DIM + vBase + lv];
      __bf16 fh = (__bf16)f;
      Bhi[lv][lk] = fh;
      Blo[lv][lk] = (__bf16)(f - (float)fh);
    }
    __syncthreads();

    // Pack fragments, same contiguous K order (k = 16*h + i) for A and B
    v16bf aHi[2], aLo[2], bHi[2], bLo[2];
#pragma unroll
    for (int mt = 0; mt < 2; ++mt) {
      int row = wS * 32 + mt * 16 + l15;
#pragma unroll
      for (int i2 = 0; i2 < 8; ++i2) {            // elems 0..15: K=h*16+i (matches B)
        aHi[mt][i2]     = Ahi[row][h * 16 + i2];
        aHi[mt][8 + i2] = Ahi[row][h * 16 + 8 + i2];
        aLo[mt][i2]     = Alo[row][h * 16 + i2];
        aLo[mt][8 + i2] = Alo[row][h * 16 + 8 + i2];
      }
    }
#pragma unroll
    for (int nt = 0; nt < 2; ++nt) {
      int col = wV * 32 + nt * 16 + l15;
#pragma unroll
      for (int i2 = 0; i2 < 16; ++i2) {           // K = h*16 + i, contiguous in LDS row
        bHi[nt][i2] = Bhi[col][h * 16 + i2];
        bLo[nt][i2] = Blo[col][h * 16 + i2];
      }
    }
#pragma unroll
    for (int mt = 0; mt < 2; ++mt)
#pragma unroll
      for (int nt = 0; nt < 2; ++nt) {
        acc[mt][nt] = __builtin_amdgcn_wmma_f32_16x16x32_bf16(false, aHi[mt], false, bHi[nt], (short)0, acc[mt][nt], false, false);
        acc[mt][nt] = __builtin_amdgcn_wmma_f32_16x16x32_bf16(false, aLo[mt], false, bHi[nt], (short)0, acc[mt][nt], false, false);
        acc[mt][nt] = __builtin_amdgcn_wmma_f32_16x16x32_bf16(false, aHi[mt], false, bLo[nt], (short)0, acc[mt][nt], false, false);
      }
    __syncthreads();
  }

  // D layout: VGPR r holds M = r + 8*h, N = lane&15
#pragma unroll
  for (int mt = 0; mt < 2; ++mt)
#pragma unroll
    for (int nt = 0; nt < 2; ++nt)
#pragma unroll
      for (int r = 0; r < 8; ++r) {
        int srow = sBase + wS * 32 + mt * 16 + h * 8 + r;
        int vcol = vBase + wV * 32 + nt * 16 + l15;
        latch_out[((size_t)b * S_DIM + srow) * V_DIM + vcol] = acc[mt][nt][r];
      }
}

// ---------------- K3: d = <sp, latch_{s-1}>, m = ||latch_{s-1}||^2 ---------
__global__ __launch_bounds__(256) void k_dm(const float* __restrict__ latch_out,
                                            const float* __restrict__ latch_init,
                                            const float* __restrict__ sp,
                                            float* __restrict__ d,
                                            float* __restrict__ m) {
  int bs = blockIdx.x, tid = threadIdx.x;
  int b = bs >> 8, s = bs & 255;
  const float* l = (s == 0) ? (latch_init + (size_t)b * V_DIM)
                            : (latch_out + ((size_t)bs - 1) * V_DIM);
  float dd = 0.f, mm = 0.f;
  for (int v = tid; v < V_DIM; v += 256) { float lv = l[v]; dd += sp[v] * lv; mm += lv * lv; }
  __shared__ float r1[256], r2[256];
  r1[tid] = dd; r2[tid] = mm; __syncthreads();
  for (int off = 128; off; off >>= 1) {
    if (tid < off) { r1[tid] += r1[tid + off]; r2[tid] += r2[tid + off]; }
    __syncthreads();
  }
  if (tid == 0) { d[bs] = r1[0]; m[bs] = r2[0]; }
}

// ---------------- K4: pointer/softmax chain, emits stack coefficients ------
__global__ void k_pointer(const float* __restrict__ d, const float* __restrict__ m,
                          const float* __restrict__ scal, const float* __restrict__ sharpen_ptr,
                          float* __restrict__ cA, float* __restrict__ cB,
                          float* __restrict__ cC, float* __restrict__ cW,
                          float* __restrict__ pops) {
  int b = blockIdx.x, n = threadIdx.x;          // 32 lanes, lanes 0..11 hold slots
  bool valid = n < N_STACK;
  float sharpen = sharpen_ptr[0];
  float nsp = scal[1];
  float pt = (n == 0) ? 1.f : 0.f;              // one-hot pointer at slot 0
  for (int s = 0; s < S_DIM; ++s) {
    float z = valid ? pt * sharpen : -1e30f;
    float mx = z;
    for (int off = 16; off; off >>= 1) mx = fmaxf(mx, __shfl_xor(mx, off, 32));
    float ex = valid ? __expf(z - mx) : 0.f;
    float sum = ex;
    for (int off = 16; off; off >>= 1) sum += __shfl_xor(sum, off, 32);
    float psh = ex / sum;
    float push_p = __shfl(psh, (n + 11) % N_STACK, 32);  // roll +1
    float pop_p  = __shfl(psh, (n + 1) % N_STACK, 32);   // roll -1
    int bs = b * S_DIM + s;
    float cs = d[bs] / (nsp * fmaxf(sqrtf(m[bs]), EPS_CS));
    float pop = cs > 0.f ? cs : expm1f(cs);              // elu
    float push = 1.f - pop;
    float pb = push * push_p, pc = pop * psh;
    float newp = push * push_p + pop * pop_p;
    if (valid) {
      int idx = bs * N_STACK + n;
      cA[idx] = 1.f - pb - pc;
      cB[idx] = pb;
      cC[idx] = pc * ZERO_OFF;
      cW[idx] = newp;
      pt = newp;
    }
    if (n == 0) pops[bs] = pop;
  }
}

// ---------------- K5: stack scan (bandwidth-bound, 131K parallel lanes) ----
__global__ __launch_bounds__(256) void k_stack(const float* __restrict__ x,
                                               const float* __restrict__ cA, const float* __restrict__ cB,
                                               const float* __restrict__ cC, const float* __restrict__ cW,
                                               const float* __restrict__ pops,
                                               float* __restrict__ outputs, float* __restrict__ tops) {
  int b = blockIdx.y;
  int v = blockIdx.x * 256 + threadIdx.x;
  float st[N_STACK];
#pragma unroll
  for (int n = 0; n < N_STACK; ++n) st[n] = ZERO_OFF;
  for (int s = 0; s < S_DIM; ++s) {
    int bs = b * S_DIM + s;                       // uniform -> scalar loads for coeffs
    float xv = x[(size_t)bs * V_DIM + v];
    float pop = pops[bs];
    float top = 0.f;
#pragma unroll
    for (int n = 0; n < N_STACK; ++n) {
      int idx = bs * N_STACK + n;
      st[n] = fmaf(st[n], cA[idx], fmaf(xv, cB[idx], cC[idx]));
      top = fmaf(st[n], cW[idx], top);
    }
    tops[(size_t)bs * V_DIM + v] = top;
    outputs[(size_t)bs * V_DIM + v] = pop * top;
  }
}

// ---------------------------------------------------------------------------
extern "C" void kernel_launch(void* const* d_in, const int* in_sizes, int n_in,
                              void* d_out, int out_size, void* d_ws, size_t ws_size,
                              hipStream_t stream) {
  const float* x            = (const float*)d_in[0];   // [B,S,V]
  const float* latch_init   = (const float*)d_in[1];   // [B,V]
  const float* should_pop   = (const float*)d_in[2];   // [V]
  const float* sharpen      = (const float*)d_in[3];   // [1]
  const float* latch_enable = (const float*)d_in[4];   // [V]

  float* out          = (float*)d_out;
  float* outputs      = out;                 // [B,S,V]
  float* latch_states = out + BSV;           // [B,S,V]
  float* pops         = out + 2 * BSV;       // [B,S]
  float* tops         = out + 2 * BSV + BS;  // [B,S,V]

  float* ws     = (float*)d_ws;
  float* ws_p   = ws;                        // [B,S]
  float* ws_nrm = ws + BS;                   // [B,S]
  float* ws_d   = ws + 2 * BS;               // [B,S]
  float* ws_m   = ws + 3 * BS;               // [B,S]
  float* ws_cp  = ws + 4 * BS;               // [B,S]
  float* ws_g   = ws + 5 * BS;               // [B,KPAD]
  float* ws_cA  = ws_g + B_DIM * KPAD;       // [B,S,12] x4
  float* ws_cB  = ws_cA + BS * N_STACK;
  float* ws_cC  = ws_cB + BS * N_STACK;
  float* ws_cW  = ws_cC + BS * N_STACK;
  float* ws_scal = ws_cW + BS * N_STACK;     // [2]

  k_norms<<<1, 256, 0, stream>>>(latch_enable, should_pop, ws_scal);
  k_row_reduce<<<BS, 256, 0, stream>>>(x, latch_enable, ws_p, ws_nrm);
  k_prefix<<<B_DIM, 256, 0, stream>>>(ws_p, ws_nrm, ws_scal, ws_cp, ws_g);
  latch_gemm_wmma<<<dim3(V_DIM / 128, S_DIM / 64, B_DIM), 256, 0, stream>>>(
      x, latch_init, ws_cp, ws_g, latch_states);
  k_dm<<<BS, 256, 0, stream>>>(latch_states, latch_init, should_pop, ws_d, ws_m);
  k_pointer<<<B_DIM, 32, 0, stream>>>(ws_d, ws_m, ws_scal, sharpen,
                                      ws_cA, ws_cB, ws_cC, ws_cW, pops);
  k_stack<<<dim3(V_DIM / 256, B_DIM), 256, 0, stream>>>(
      x, ws_cA, ws_cB, ws_cC, ws_cW, pops, outputs, tops);
}
